// RecurrentHolE_89077621719316
// MI455X (gfx1250) — matrix operand order repaired and checked by reading, not compiled, not run on hardware
//
#include <hip/hip_runtime.h>
#include <hip/hip_bf16.h>

typedef __attribute__((ext_vector_type(16))) _Float16 v16h;
typedef __attribute__((ext_vector_type(8)))  float    v8f;
typedef __attribute__((ext_vector_type(4)))  float    f32x4;

#define EN 40000
#define RN 500
#define DD 128
#define HH 8
#define DHH 16
#define FF 512
#define BB 16
#define SS 4
#define OUT_HIST (EN * BB)

// ---------------- kernel 1: gather embeddings into x[s][b][d] ----------------
__global__ void k_embed(const int* __restrict__ triple, const float* __restrict__ ent,
                        const float* __restrict__ rel, const float* __restrict__ ctx,
                        float* __restrict__ x) {
  for (int i = threadIdx.x; i < SS * BB * DD; i += 256) {
    int s = i >> 11;            // / (16*128)
    int rem = i & 2047;
    int b = rem >> 7, d = rem & 127;
    float v;
    if (s == 0)      v = ent[triple[0 * BB + b] * DD + d];
    else if (s == 1) v = rel[triple[1 * BB + b] * DD + d];
    else if (s == 2) v = ent[triple[2 * BB + b] * DD + d];
    else             v = ctx[b * DD + d];
    x[i] = v;
  }
}

// ---------------- kernel 2: attention half-layer (one block per batch b) -----
__global__ void k_attn(float* __restrict__ x, const float* __restrict__ wi,
                       const float* __restrict__ bi, const float* __restrict__ wo,
                       const float* __restrict__ bo, const float* __restrict__ g1,
                       const float* __restrict__ b1) {
  __shared__ float xs[SS][DD];
  __shared__ float qkv[SS][3 * DD];
  __shared__ float o[SS][DD];
  __shared__ float r[SS][DD];
  __shared__ float mstat[SS], vstat[SS];
  int b = blockIdx.x, tid = threadIdx.x;

  for (int i = tid; i < SS * DD; i += 256)
    xs[i >> 7][i & 127] = x[((i >> 7) * BB + b) * DD + (i & 127)];
  __syncthreads();

  // qkv projection: 4 tokens x 384 outputs, 128-dot each
  for (int i = tid; i < SS * 3 * DD; i += 256) {
    int t = i / (3 * DD), e = i % (3 * DD);
    const float* wr = wi + e * DD;
    float acc = bi[e];
    for (int d = 0; d < DD; ++d) acc += xs[t][d] * wr[d];
    qkv[t][e] = acc;
  }
  __syncthreads();

  // attention: one thread per (head, s) pair
  if (tid < HH * SS) {
    int h = tid >> 2, s = tid & 3;
    float logits[SS];
    float mx = -1e30f;
    for (int t = 0; t < SS; ++t) {
      float a = 0.f;
      for (int d = 0; d < DHH; ++d)
        a += qkv[s][h * DHH + d] * qkv[t][DD + h * DHH + d];
      a *= 0.25f;  // 1/sqrt(16)
      logits[t] = a;
      mx = fmaxf(mx, a);
    }
    float den = 0.f;
    for (int t = 0; t < SS; ++t) { logits[t] = __expf(logits[t] - mx); den += logits[t]; }
    float inv = 1.0f / den;
    for (int d = 0; d < DHH; ++d) {
      float acc = 0.f;
      for (int t = 0; t < SS; ++t) acc += logits[t] * qkv[t][2 * DD + h * DHH + d];
      o[s][h * DHH + d] = acc * inv;
    }
  }
  __syncthreads();

  // out projection + residual
  for (int i = tid; i < SS * DD; i += 256) {
    int t = i >> 7, e = i & 127;
    const float* wr = wo + e * DD;
    float acc = bo[e];
    for (int d = 0; d < DD; ++d) acc += o[t][d] * wr[d];
    r[t][e] = xs[t][e] + acc;
  }
  __syncthreads();
  if (tid < SS) {
    float m = 0.f;
    for (int d = 0; d < DD; ++d) m += r[tid][d];
    m *= (1.0f / DD);
    float vv = 0.f;
    for (int d = 0; d < DD; ++d) { float df = r[tid][d] - m; vv += df * df; }
    mstat[tid] = m;
    vstat[tid] = vv * (1.0f / DD);
  }
  __syncthreads();
  for (int i = tid; i < SS * DD; i += 256) {
    int t = i >> 7, d = i & 127;
    float y = (r[t][d] - mstat[t]) * rsqrtf(vstat[t] + 1e-5f) * g1[d] + b1[d];
    x[(t * BB + b) * DD + d] = y;
  }
}

// ---------------- kernel 3: feed-forward half-layer (one block per token) ----
__global__ void k_ff(float* __restrict__ x, const float* __restrict__ w1,
                     const float* __restrict__ bf1, const float* __restrict__ w2,
                     const float* __restrict__ bf2, const float* __restrict__ g2,
                     const float* __restrict__ b2l) {
  __shared__ float xs[DD];
  __shared__ float hbuf[FF];
  __shared__ float r[DD];
  __shared__ float ms, vs;
  int tok = blockIdx.x;  // s*16+b
  int tid = threadIdx.x;
  if (tid < DD) xs[tid] = x[tok * DD + tid];
  __syncthreads();
  for (int i = tid; i < FF; i += 256) {
    const float* wr = w1 + i * DD;
    float acc = bf1[i];
    for (int d = 0; d < DD; ++d) acc += xs[d] * wr[d];
    hbuf[i] = fmaxf(acc, 0.f);
  }
  __syncthreads();
  if (tid < DD) {
    const float* wr = w2 + tid * FF;
    float acc = bf2[tid];
    for (int f = 0; f < FF; ++f) acc += hbuf[f] * wr[f];
    r[tid] = xs[tid] + acc;
  }
  __syncthreads();
  if (tid == 0) {
    float m = 0.f;
    for (int d = 0; d < DD; ++d) m += r[d];
    m *= (1.0f / DD);
    float vv = 0.f;
    for (int d = 0; d < DD; ++d) { float df = r[d] - m; vv += df * df; }
    ms = m;
    vs = vv * (1.0f / DD);
  }
  __syncthreads();
  if (tid < DD)
    x[tok * DD + tid] = (r[tid] - ms) * rsqrtf(vs + 1e-5f) * g2[tid] + b2l[tid];
}

// ---------------- kernel 4: circular conv w[b][j] + hist output --------------
__global__ void k_conv(const float* __restrict__ x, const int* __restrict__ next_triple,
                       const float* __restrict__ rel, float* __restrict__ wconv,
                       float* __restrict__ out) {
  __shared__ float subj[BB][DD];
  __shared__ float nr[BB][DD];
  int tid = threadIdx.x;
  for (int i = tid; i < BB * DD; i += 256) {
    int b = i >> 7, d = i & 127;
    subj[b][d] = x[(0 * BB + b) * DD + d];
    nr[b][d] = rel[next_triple[1 * BB + b] * DD + d];
    out[OUT_HIST + i] = x[(3 * BB + b) * DD + d];  // hist = x[S-1]
  }
  __syncthreads();
  for (int i = tid; i < BB * DD; i += 256) {
    int b = i >> 7, j = i & 127;
    float acc = 0.f;
    for (int k = 0; k < DD; ++k) acc += subj[b][k] * nr[b][(j - k) & 127];
    wconv[i] = acc;
  }
}

// ---------------- kernel 5: scores = ent(40000x128) x w^T(128x16), WMMA ------
// grid = 625 blocks x 128 threads (4 waves); each wave owns a 16-entity tile.
__global__ void __launch_bounds__(128) k_score(const float* __restrict__ ent,
                                               const float* __restrict__ wconv,
                                               float* __restrict__ out) {
  __shared__ float st[4][16][16];
  int lane = threadIdx.x & 31;
  int wv = threadIdx.x >> 5;
  int e0 = (blockIdx.x * 4 + wv) * 16;
  int m = lane & 15;      // entity row within tile (A) / batch col (B,C,D)
  int hf = lane >> 4;

  // B operand: w^T, 32x16 f16 per k-tile; lane needs w[n=m][kbase+8*hf+0..7] and +16
  v16h Bv[4];
  const float* wrow = wconv + m * DD + hf * 8;
#pragma unroll
  for (int kt = 0; kt < 4; ++kt) {
    f32x4 b0 = *(const f32x4*)(wrow + kt * 32);
    f32x4 b1 = *(const f32x4*)(wrow + kt * 32 + 4);
    f32x4 b2 = *(const f32x4*)(wrow + kt * 32 + 16);
    f32x4 b3 = *(const f32x4*)(wrow + kt * 32 + 20);
#pragma unroll
    for (int j = 0; j < 4; ++j) {
      Bv[kt][j]      = (_Float16)b0[j];
      Bv[kt][4 + j]  = (_Float16)b1[j];
      Bv[kt][8 + j]  = (_Float16)b2[j];
      Bv[kt][12 + j] = (_Float16)b3[j];
    }
  }

  // A operand: 16x32 f16 per k-tile; lane needs ent[e0+m][kbase+8*hf+0..7] and +16
  v8f acc = {};
  const float* arow = ent + (size_t)(e0 + m) * DD + hf * 8;
#pragma unroll
  for (int kt = 0; kt < 4; ++kt) {
    f32x4 a0 = *(const f32x4*)(arow + kt * 32);
    f32x4 a1 = *(const f32x4*)(arow + kt * 32 + 4);
    f32x4 a2 = *(const f32x4*)(arow + kt * 32 + 16);
    f32x4 a3 = *(const f32x4*)(arow + kt * 32 + 20);
    v16h A;
#pragma unroll
    for (int j = 0; j < 4; ++j) {
      A[j]      = (_Float16)a0[j];
      A[4 + j]  = (_Float16)a1[j];
      A[8 + j]  = (_Float16)a2[j];
      A[12 + j] = (_Float16)a3[j];
    }
    acc = __builtin_amdgcn_wmma_f32_16x16x32_f16(false, A, false, Bv[kt],
                                                 (short)0, acc, false, false);
  }

  // D layout: VGPR j, lane l -> scores[n = l%16][m' = j + 8*(l/16)].
  // Transpose through LDS so stores are 64B-contiguous per batch row.
#pragma unroll
  for (int j = 0; j < 8; ++j) st[wv][m][j + 8 * hf] = acc[j];
  __syncthreads();
  int nn = lane >> 1, mc = (lane & 1) * 8;
  const float* src = &st[wv][nn][mc];
  f32x4 r0 = *(const f32x4*)src;
  f32x4 r1 = *(const f32x4*)(src + 4);
  float* dst = out + (size_t)nn * EN + e0 + mc;
  *(f32x4*)dst = r0;
  *(f32x4*)(dst + 4) = r1;
}

extern "C" void kernel_launch(void* const* d_in, const int* in_sizes, int n_in,
                              void* d_out, int out_size, void* d_ws, size_t ws_size,
                              hipStream_t stream) {
  const int*   triple      = (const int*)d_in[0];
  const int*   next_triple = (const int*)d_in[1];
  const float* ctx         = (const float*)d_in[2];
  const float* ent         = (const float*)d_in[3];
  const float* rel         = (const float*)d_in[4];
  const float* in_proj_w   = (const float*)d_in[5];
  const float* in_proj_b   = (const float*)d_in[6];
  const float* out_w       = (const float*)d_in[7];
  const float* out_b       = (const float*)d_in[8];
  const float* ln1_w       = (const float*)d_in[9];
  const float* ln1_b       = (const float*)d_in[10];
  const float* ln2_w       = (const float*)d_in[11];
  const float* ln2_b       = (const float*)d_in[12];
  const float* ff1_w       = (const float*)d_in[13];
  const float* ff1_b       = (const float*)d_in[14];
  const float* ff2_w       = (const float*)d_in[15];
  const float* ff2_b       = (const float*)d_in[16];
  float* out = (float*)d_out;

  float* xbuf  = (float*)d_ws;                   // 4*16*128 floats
  float* wconv = xbuf + SS * BB * DD;            // 16*128 floats

  k_embed<<<1, 256, 0, stream>>>(triple, ent, rel, ctx, xbuf);
  for (int l = 0; l < 2; ++l) {
    k_attn<<<BB, 256, 0, stream>>>(xbuf,
        in_proj_w + (size_t)l * 3 * DD * DD, in_proj_b + (size_t)l * 3 * DD,
        out_w + (size_t)l * DD * DD, out_b + (size_t)l * DD,
        ln1_w + (size_t)l * DD, ln1_b + (size_t)l * DD);
    k_ff<<<SS * BB, 256, 0, stream>>>(xbuf,
        ff1_w + (size_t)l * FF * DD, ff1_b + (size_t)l * FF,
        ff2_w + (size_t)l * DD * FF, ff2_b + (size_t)l * DD,
        ln2_w + (size_t)l * DD, ln2_b + (size_t)l * DD);
  }
  k_conv<<<1, 256, 0, stream>>>(xbuf, next_triple, rel, wconv, out);
  k_score<<<EN / 64, 128, 0, stream>>>(ent, wconv, out);
}
